// PromptGenerator_18665927868523
// MI455X (gfx1250) — matrix operand order repaired and checked by reading, not compiled, not run on hardware
//
#include <hip/hip_runtime.h>
#include <hip/hip_bf16.h>

// ---------------------------------------------------------------------------
// Sizes from the reference
// ---------------------------------------------------------------------------
#define NUM_CQ      16
#define NUM_CODES   8192
#define NUM_SAMPLES 16
#define NHEADS      4
#define DM          32
#define DH          8
#define EMB         768
#define BB          64
#define LL          512
#define NROWS       (BB * LL)        // 32768
#define NSEL        (BB * NUM_CQ)    // 1024

// B tile staging geometry: 32 N-rows x 32 K bf16, row stride padded to 80 B
#define BROWS       32
#define BSTRIDE     80
#define BBUF        (BROWS * BSTRIDE)   // 2560 bytes per buffer

// ---------------------------------------------------------------------------
// WMMA types (CDNA5 / gfx1250, wave32)
// ---------------------------------------------------------------------------
typedef __attribute__((ext_vector_type(8)))  float  v8f;
typedef __attribute__((ext_vector_type(16))) __bf16 v16bf;

union U16x16 { v16bf v; uint4 q[2]; unsigned short u[16]; };

__device__ __forceinline__ unsigned short f2bf(float f) {
    unsigned u = __float_as_uint(f);
    u += 0x7FFFu + ((u >> 16) & 1u);   // round-to-nearest-even
    return (unsigned short)(u >> 16);
}

// LDS byte offset of a generic pointer known to point into LDS.
typedef __attribute__((address_space(3))) const void lds_cvoid;
__device__ __forceinline__ unsigned lds_offset(const void* p) {
    return (unsigned)(size_t)((lds_cvoid*)p);
}

// CDNA5 async global->LDS copy (ASYNCcnt path); portable inline asm.
__device__ __forceinline__ void async_load_b128(unsigned lds_off, const void* gptr) {
    asm volatile("global_load_async_to_lds_b128 %0, %1, off"
                 :: "v"(lds_off), "v"(gptr) : "memory");
}
__device__ __forceinline__ void wait_asynccnt0() {
    asm volatile("s_wait_asynccnt 0x0" ::: "memory");
}

// ---------------------------------------------------------------------------
// A-fragment loaders (CDNA5 ISA §7.12.2, 16-bit A 16x32):
//   lane l holds row M=l%16; elems 0..7 -> K=kofs+0..7, elems 8..15 ->
//   K=kofs+16..23, kofs=(l>=16)?8:0.
// ---------------------------------------------------------------------------
__device__ __forceinline__ v16bf load_a_frag(const unsigned short* aRow, int kc, int half) {
    U16x16 a;
    const uint4* p = reinterpret_cast<const uint4*>(aRow + kc + half * 8);
    a.q[0] = p[0];
    a.q[1] = p[2];     // +16 elements
    return a.v;
}
__device__ __forceinline__ v16bf load_a_frag(const float* aRow, int kc, int half) {
    const float4* ap = reinterpret_cast<const float4*>(aRow + kc + half * 8);
    float4 a0 = ap[0], a1 = ap[1], a2 = ap[4], a3 = ap[5];
    const float af[16] = { a0.x,a0.y,a0.z,a0.w, a1.x,a1.y,a1.z,a1.w,
                           a2.x,a2.y,a2.z,a2.w, a3.x,a3.y,a3.z,a3.w };
    U16x16 a;
    #pragma unroll
    for (int i = 0; i < 16; ++i) a.u[i] = f2bf(af[i]);
    return a.v;
}

// ---------------------------------------------------------------------------
// out = A[M,K] @ W[N,K]^T + bias via v_wmma_f32_16x16x32_bf16.
// Block = 256 threads = 8 waves; wave w owns rows [blk.x*128+w*16, +16) and a
// 16x32 output tile (2 accumulators sharing one A fragment). The 32x32 bf16
// W tile is staged into double-buffered LDS with global_load_async_to_lds_b128
// by waves 0-3 (wave-uniform predicate, EXEC stays full): iteration i waits on
// the copy issued at i-1, so async latency overlaps the previous WMMAs, and a
// single s_barrier per iteration both publishes buf[cur] and fences the reads
// of buf[cur^1] before it is overwritten (issue is after the barrier, reads
// complete before it because the WMMAs consumed them).
// B per lane: col N=l%16, elems 0..7 -> K=kofs+0..7, elems 8..15 -> K=kofs+16..23, kofs=(l>=16)?8:0 => two ds_load_b128.
// C: vreg r -> M=r+8*(l>=16), N=l%16.
// M % 128 == 0, N % 32 == 0, K % 32 == 0 at every call site.
// ---------------------------------------------------------------------------
template <typename TA>
__global__ void __launch_bounds__(256)
gemm_wmma(const TA* __restrict__ A, const unsigned short* __restrict__ W,
          const float* __restrict__ bias, float* __restrict__ C,
          unsigned short* __restrict__ Cbf, int M, int N, int K, int relu) {
    __shared__ unsigned short sB[2 * BBUF / 2];   // 2 x 2560 bytes
    const int tid  = threadIdx.x;
    const int wave = tid >> 5;
    const int lane = tid & 31;
    const int half = lane >> 4;
    const int l16  = lane & 15;
    const int rowBase = blockIdx.x * 128 + wave * 16;
    const int colBase = blockIdx.y * 32;

    const TA* aRow = A + (size_t)(rowBase + l16) * K;
    const unsigned sBoff = lds_offset(sB);
    const int rowN = tid >> 2, seg = tid & 3;   // staging role (waves 0-3)

    v8f acc0 = {}, acc1 = {};
    if (tid < 128) {       // prologue: stage first B tile into buffer 0
        async_load_b128(sBoff + rowN * BSTRIDE + seg * 16,
                        (const char*)(W + (size_t)(colBase + rowN) * K) + seg * 16);
    }

    for (int kc = 0; kc < K; kc += 32) {
        const unsigned cur = ((unsigned)kc >> 5) & 1u;
        wait_asynccnt0();      // buf[cur] complete (no-op for waves 4-7)
        __syncthreads();       // publish buf[cur]; fence prior reads of buf[cur^1]
        if (kc + 32 < K) {
            if (tid < 128) {
                const char* g = (const char*)(W + (size_t)(colBase + rowN) * K
                                              + kc + 32) + seg * 16;
                async_load_b128(sBoff + (cur ^ 1u) * BBUF + rowN * BSTRIDE + seg * 16, g);
            }
            __builtin_prefetch(aRow + kc + 32, 0, 3);   // WGP-scope A prefetch
        }

        v16bf a = load_a_frag(aRow, kc, half);
        const char* sbase = (const char*)sB + cur * BBUF;
        U16x16 b0, b1;
        const uint4* bp0 = reinterpret_cast<const uint4*>(sbase + l16 * BSTRIDE + half * 16);
        const uint4* bp1 = reinterpret_cast<const uint4*>(sbase + (16 + l16) * BSTRIDE + half * 16);
        b0.q[0] = bp0[0]; b0.q[1] = bp0[2];
        b1.q[0] = bp1[0]; b1.q[1] = bp1[2];

        acc0 = __builtin_amdgcn_wmma_f32_16x16x32_bf16(
                   false, a, false, b0.v, (short)0, acc0, false, false);
        acc1 = __builtin_amdgcn_wmma_f32_16x16x32_bf16(
                   false, a, false, b1.v, (short)0, acc1, false, false);
    }

    const float bb0 = bias ? bias[colBase + l16]      : 0.0f;
    const float bb1 = bias ? bias[colBase + 16 + l16] : 0.0f;
    #pragma unroll
    for (int r = 0; r < 8; ++r) {
        float v0 = acc0[r] + bb0;
        float v1 = acc1[r] + bb1;
        if (relu) { v0 = fmaxf(v0, 0.0f); v1 = fmaxf(v1, 0.0f); }
        const size_t idx = (size_t)(rowBase + r + half * 8) * N + colBase + l16;
        if (C)   { C[idx]   = v0;       C[idx + 16]   = v1; }
        if (Cbf) { Cbf[idx] = f2bf(v0); Cbf[idx + 16] = f2bf(v1); }
    }
}

// fp32 -> packed bf16 converter (weights + codebook, once per launch)
__global__ void __launch_bounds__(256)
cvt_bf16_kernel(const float* __restrict__ in, unsigned short* __restrict__ out, int n) {
    const int i = blockIdx.x * 256 + threadIdx.x;
    if (i < n) out[i] = f2bf(in[i]);
}

// ---------------------------------------------------------------------------
// Fused attention for one (batch, head): online-softmax over 512 keys, dh=8.
// K/V staged in LDS (32 KB + 2 KB mask). Output feeds only the O-projection
// GEMM, so it is written as bf16 directly.
// ---------------------------------------------------------------------------
__global__ void __launch_bounds__(256)
attn_kernel(const float* __restrict__ qkv, const int* __restrict__ mask,
            unsigned short* __restrict__ xatt_bf) {
    __shared__ float kLds[LL][DH];
    __shared__ float vLds[LL][DH];
    __shared__ float neg[LL];
    const int b = blockIdx.x, h = blockIdx.y;
    const int tid = threadIdx.x;

    for (int j = tid; j < LL; j += 256) {
        const float* base = qkv + (size_t)(b * LL + j) * (3 * DM);
        #pragma unroll
        for (int d = 0; d < DH; ++d) {
            kLds[j][d] = base[DM     + h * DH + d];
            vLds[j][d] = base[2 * DM + h * DH + d];
        }
        neg[j] = (j < NUM_CQ || mask[b * (LL - NUM_CQ) + (j - NUM_CQ)] != 0)
                     ? 0.0f : -1.0e9f;
    }
    __syncthreads();

    const float scale = 0.3535533906f;  // 1/sqrt(8)
    for (int q = tid; q < LL; q += 256) {
        const float* base = qkv + (size_t)(b * LL + q) * (3 * DM) + h * DH;
        float qv[DH];
        #pragma unroll
        for (int d = 0; d < DH; ++d) qv[d] = base[d];
        float m = -3.0e38f, l = 0.0f, acc[DH] = {};
        for (int j = 0; j < LL; ++j) {
            float s = 0.0f;
            #pragma unroll
            for (int d = 0; d < DH; ++d) s += qv[d] * kLds[j][d];
            s = s * scale + neg[j];
            float mNew = fmaxf(m, s);
            float corr = __expf(m - mNew);
            float p    = __expf(s - mNew);
            l = l * corr + p;
            #pragma unroll
            for (int d = 0; d < DH; ++d) acc[d] = acc[d] * corr + p * vLds[j][d];
            m = mNew;
        }
        const float inv = 1.0f / l;
        unsigned short* o = xatt_bf + (size_t)(b * LL + q) * DM + h * DH;
        #pragma unroll
        for (int d = 0; d < DH; ++d) o[d] = f2bf(acc[d] * inv);
    }
}

// ---------------------------------------------------------------------------
// x = LayerNorm(x + o); dual-writes fp32 (residual stream) + bf16 (GEMM feed)
// ---------------------------------------------------------------------------
__global__ void __launch_bounds__(256)
add_ln_kernel(float* __restrict__ x, unsigned short* __restrict__ xbf,
              const float* __restrict__ o,
              const float* __restrict__ g, const float* __restrict__ bt, int M) {
    const int row = blockIdx.x * blockDim.x + threadIdx.x;
    if (row >= M) return;
    float v[DM];
    float* xr = x + (size_t)row * DM;
    unsigned short* xb = xbf + (size_t)row * DM;
    const float* orr = o + (size_t)row * DM;
    float mu = 0.0f;
    #pragma unroll
    for (int i = 0; i < DM; ++i) { v[i] = xr[i] + orr[i]; mu += v[i]; }
    mu *= (1.0f / DM);
    float var = 0.0f;
    #pragma unroll
    for (int i = 0; i < DM; ++i) { float d = v[i] - mu; var += d * d; }
    var *= (1.0f / DM);
    const float rstd = rsqrtf(var + 1e-5f);
    #pragma unroll
    for (int i = 0; i < DM; ++i) {
        const float y = g[i] * (v[i] - mu) * rstd + bt[i];
        xr[i] = y;
        xb[i] = f2bf(y);
    }
}

// Gather x[:, :NUM_CQ, :] -> xsel_bf[1024, 32] (bf16, feeds fc_out GEMM)
__global__ void __launch_bounds__(256)
gather_cq_kernel(const float* __restrict__ x, unsigned short* __restrict__ xsel) {
    const int gidx = blockIdx.x * 256 + threadIdx.x;
    if (gidx >= NSEL * DM) return;
    const int n = gidx >> 5, c = gidx & 31;
    const int b = n >> 4, i = n & 15;
    xsel[gidx] = f2bf(x[((size_t)(b * LL + i)) * DM + c]);
}

// Squared L2 norm per row; one block per row.
__global__ void __launch_bounds__(256)
rownorm_kernel(const float* __restrict__ X, float* __restrict__ out, int K) {
    __shared__ float red[256];
    const float* p = X + (size_t)blockIdx.x * K;
    float s = 0.0f;
    for (int i = threadIdx.x; i < K; i += 256) { float v = p[i]; s += v * v; }
    red[threadIdx.x] = s;
    __syncthreads();
    for (int off = 128; off; off >>= 1) {
        if (threadIdx.x < off) red[threadIdx.x] += red[threadIdx.x + off];
        __syncthreads();
    }
    if (threadIdx.x == 0) out[blockIdx.x] = red[0];
}

// ---------------------------------------------------------------------------
// Gumbel-max multinomial (deterministic hash RNG) + codebook average.
// xc_quantized == xq numerically (straight-through estimator).
// ---------------------------------------------------------------------------
__device__ __forceinline__ float uhash01(unsigned x) {
    x ^= x >> 17; x *= 0xED5AD4BBu;
    x ^= x >> 11; x *= 0xAC4C1B51u;
    x ^= x >> 15; x *= 0x31848BABu;
    x ^= x >> 14;
    return (float)(x >> 8) * (1.0f / 16777216.0f) + 1.0e-7f;
}

__global__ void __launch_bounds__(256)
sample_kernel(const float* __restrict__ dot, const float* __restrict__ rn,
              const float* __restrict__ cn, const float* __restrict__ cb,
              float* __restrict__ out) {
    __shared__ float bestV[256];
    __shared__ int   bestI[256];
    __shared__ int   draws[NUM_SAMPLES];
    const int n = blockIdx.x;
    const float* drow = dot + (size_t)n * NUM_CODES;
    const float r = rn[n];

    for (int s = 0; s < NUM_SAMPLES; ++s) {
        float bv = -3.4e38f; int bi = 0;
        for (int j = threadIdx.x; j < NUM_CODES; j += 256) {
            const float logit = -(r + cn[j] - 2.0f * drow[j]) - 1e-5f;
            const float u = uhash01((unsigned)(n * NUM_SAMPLES + s) * 0x9E3779B9u
                                    ^ (unsigned)j * 0x85EBCA6Bu);
            const float v = logit - __logf(-__logf(u));   // Gumbel-max
            if (v > bv) { bv = v; bi = j; }
        }
        bestV[threadIdx.x] = bv; bestI[threadIdx.x] = bi;
        __syncthreads();
        for (int off = 128; off; off >>= 1) {
            if (threadIdx.x < off && bestV[threadIdx.x + off] > bestV[threadIdx.x]) {
                bestV[threadIdx.x] = bestV[threadIdx.x + off];
                bestI[threadIdx.x] = bestI[threadIdx.x + off];
            }
            __syncthreads();
        }
        if (threadIdx.x == 0) draws[s] = bestI[0];
        __syncthreads();
    }

    for (int e = threadIdx.x; e < EMB; e += 256) {
        float sum = 0.0f;
        #pragma unroll
        for (int s = 0; s < NUM_SAMPLES; ++s) sum += cb[(size_t)draws[s] * EMB + e];
        out[(size_t)n * EMB + e] = sum * (1.0f / NUM_SAMPLES);
    }
}

// ---------------------------------------------------------------------------
// Host-side pipeline
// ---------------------------------------------------------------------------
extern "C" void kernel_launch(void* const* d_in, const int* in_sizes, int n_in,
                              void* d_out, int out_size, void* d_ws, size_t ws_size,
                              hipStream_t stream) {
    const float* xcq     = (const float*)d_in[0];
    const int*   amask   = (const int*)  d_in[1];
    const float* fc_in_w = (const float*)d_in[2];
    const float* fc_in_b = (const float*)d_in[3];
    const float* fc_ow   = (const float*)d_in[4];
    const float* fc_ob   = (const float*)d_in[5];
    const float* in_w    = (const float*)d_in[6];
    const float* in_b    = (const float*)d_in[7];
    const float* out_w   = (const float*)d_in[8];
    const float* out_b   = (const float*)d_in[9];
    const float* ff1_w   = (const float*)d_in[10];
    const float* ff1_b   = (const float*)d_in[11];
    const float* ff2_w   = (const float*)d_in[12];
    const float* ff2_b   = (const float*)d_in[13];
    const float* ln1_g   = (const float*)d_in[14];
    const float* ln1_b   = (const float*)d_in[15];
    const float* ln2_g   = (const float*)d_in[16];
    const float* ln2_b   = (const float*)d_in[17];
    const float* cbk     = (const float*)d_in[18];

    // Workspace carving (~77 MB, all regions 256B-aligned sizes)
    char* p = (char*)d_ws;
    float*          x      = (float*)p;          p += (size_t)NROWS * DM * 4;        // 4 MB
    unsigned short* xbf    = (unsigned short*)p; p += (size_t)NROWS * DM * 2;        // 2 MB
    float*          qkv    = (float*)p;          p += (size_t)NROWS * 3 * DM * 4;    // 12 MB
    unsigned short* xattb  = (unsigned short*)p; p += (size_t)NROWS * DM * 2;        // 2 MB
    float*          xo     = (float*)p;          p += (size_t)NROWS * DM * 4;        // 4 MB
    unsigned short* h1b    = (unsigned short*)p; p += (size_t)NROWS * 2 * DM * 2;    // 4 MB
    unsigned short* xselb  = (unsigned short*)p; p += (size_t)NSEL * DM * 2;
    float*          flat   = (float*)p;          p += (size_t)NSEL * EMB * 4;        // 3 MB
    unsigned short* flatb  = (unsigned short*)p; p += (size_t)NSEL * EMB * 2;        // 1.5 MB
    float*          dotb   = (float*)p;          p += (size_t)NSEL * NUM_CODES * 4;  // 32 MB
    float*          rn     = (float*)p;          p += (size_t)NSEL * 4;
    float*          cn     = (float*)p;          p += (size_t)NUM_CODES * 4;
    unsigned short* fciwb  = (unsigned short*)p; p += (size_t)DM * EMB * 2;
    unsigned short* fcowb  = (unsigned short*)p; p += (size_t)EMB * DM * 2;
    unsigned short* inwb   = (unsigned short*)p; p += (size_t)2 * 3 * DM * DM * 2;
    unsigned short* outwb  = (unsigned short*)p; p += (size_t)2 * DM * DM * 2;
    unsigned short* ff1wb  = (unsigned short*)p; p += (size_t)2 * 2 * DM * DM * 2;
    unsigned short* ff2wb  = (unsigned short*)p; p += (size_t)2 * DM * 2 * DM * 2;
    unsigned short* cbkb   = (unsigned short*)p; p += (size_t)NUM_CODES * EMB * 2;   // 12 MB

    const dim3 blk(256);
    #define CVT(src, dst, n) \
        cvt_bf16_kernel<<<((n) + 255) / 256, blk, 0, stream>>>(src, dst, n)

    // One-time weight conversions (bf16 operands for all WMMA GEMMs)
    CVT(fc_in_w, fciwb, DM * EMB);
    CVT(fc_ow,   fcowb, EMB * DM);
    CVT(in_w,    inwb,  2 * 3 * DM * DM);
    CVT(out_w,   outwb, 2 * DM * DM);
    CVT(ff1_w,   ff1wb, 2 * 2 * DM * DM);
    CVT(ff2_w,   ff2wb, 2 * DM * 2 * DM);
    CVT(cbk,     cbkb,  NUM_CODES * EMB);

    // x = xcq @ fc_in_w^T + b   (memory-bound: streams 100 MB of xcq once)
    gemm_wmma<float><<<dim3(NROWS / 128, DM / 32), blk, 0, stream>>>(
        xcq, fciwb, fc_in_b, x, xbf, NROWS, DM, EMB, 0);

    for (int l = 0; l < 2; ++l) {
        gemm_wmma<unsigned short><<<dim3(NROWS / 128, (3 * DM) / 32), blk, 0, stream>>>(
            xbf, inwb + (size_t)l * 3 * DM * DM, in_b + l * 3 * DM,
            qkv, nullptr, NROWS, 3 * DM, DM, 0);
        attn_kernel<<<dim3(BB, NHEADS), blk, 0, stream>>>(qkv, amask, xattb);
        gemm_wmma<unsigned short><<<dim3(NROWS / 128, DM / 32), blk, 0, stream>>>(
            xattb, outwb + (size_t)l * DM * DM, out_b + l * DM,
            xo, nullptr, NROWS, DM, DM, 0);
        add_ln_kernel<<<NROWS / 256, blk, 0, stream>>>(
            x, xbf, xo, ln1_g + l * DM, ln1_b + l * DM, NROWS);
        gemm_wmma<unsigned short><<<dim3(NROWS / 128, (2 * DM) / 32), blk, 0, stream>>>(
            xbf, ff1wb + (size_t)l * 2 * DM * DM, ff1_b + l * 2 * DM,
            nullptr, h1b, NROWS, 2 * DM, DM, 1);
        gemm_wmma<unsigned short><<<dim3(NROWS / 128, DM / 32), blk, 0, stream>>>(
            h1b, ff2wb + (size_t)l * DM * 2 * DM, ff2_b + l * DM,
            xo, nullptr, NROWS, DM, 2 * DM, 0);
        add_ln_kernel<<<NROWS / 256, blk, 0, stream>>>(
            x, xbf, xo, ln2_g + l * DM, ln2_b + l * DM, NROWS);
    }

    // xc_out = x[:, :16] @ fc_out_w^T + b
    gather_cq_kernel<<<(NSEL * DM + 255) / 256, blk, 0, stream>>>(x, xselb);
    gemm_wmma<unsigned short><<<dim3(NSEL / 128, EMB / 32), blk, 0, stream>>>(
        xselb, fcowb, fc_ob, flat, flatb, NSEL, EMB, DM, 0);

    // Squared-distance pieces + codebook GEMM (12.9 GFLOP, the WMMA workload)
    rownorm_kernel<<<NSEL, blk, 0, stream>>>(flat, rn, EMB);
    rownorm_kernel<<<NUM_CODES, blk, 0, stream>>>(cbk, cn, EMB);
    gemm_wmma<unsigned short><<<dim3(NSEL / 128, NUM_CODES / 32), blk, 0, stream>>>(
        flatb, cbkb, nullptr, dotb, nullptr, NSEL, NUM_CODES, EMB, 0);

    // Multinomial sampling + codebook average -> output tuple element 0
    sample_kernel<<<NSEL, blk, 0, stream>>>(dotb, rn, cn, cbk, (float*)d_out);

    // Tuple element 1 is xcq verbatim (200 MB HBM traffic = the real floor)
    hipMemcpyAsync((float*)d_out + (size_t)NSEL * EMB, xcq,
                   (size_t)NROWS * EMB * sizeof(float),
                   hipMemcpyDeviceToDevice, stream);
}
